// GRU_my_9285719294338
// MI455X (gfx1250) — compile-verified, matrix operand order checked by reading, not run on hardware
//
#include <hip/hip_runtime.h>

typedef __bf16 bf16_t;
typedef __attribute__((ext_vector_type(16))) __bf16 v16bf;
typedef __attribute__((ext_vector_type(8)))  float  v8f;

#define LDSP 40   // padded LDS row stride in bf16 (32 data + 8 pad), keeps 16B alignment

union FragU { uint4 u[2]; v16bf v; };

static __device__ __forceinline__ float sigmoidf_(float x) {
  return 1.0f / (1.0f + __expf(-x));
}
static __device__ __forceinline__ float tanhf_(float x) {
  return 2.0f / (1.0f + __expf(-2.0f * x)) - 1.0f;
}

// ---------------------------------------------------------------------------
// f32 -> bf16 bulk convert, 8 elements / thread
// ---------------------------------------------------------------------------
__global__ void cvt_f32_bf16(const float* __restrict__ in, bf16_t* __restrict__ out,
                             long long n8) {
  long long i = blockIdx.x * (long long)blockDim.x + threadIdx.x;
  if (i >= n8) return;
  const float4* p = (const float4*)in + 2 * i;
  float4 a = p[0], b = p[1];
  union { bf16_t h[8]; uint4 u; } t;
  t.h[0] = (bf16_t)a.x; t.h[1] = (bf16_t)a.y; t.h[2] = (bf16_t)a.z; t.h[3] = (bf16_t)a.w;
  t.h[4] = (bf16_t)b.x; t.h[5] = (bf16_t)b.y; t.h[6] = (bf16_t)b.z; t.h[7] = (bf16_t)b.w;
  ((uint4*)out)[i] = t.u;
}

__global__ void copy_f32(const float* __restrict__ in, float* __restrict__ out, int n) {
  int i = blockIdx.x * blockDim.x + threadIdx.x;
  if (i < n) out[i] = in[i];
}

// ---------------------------------------------------------------------------
// Input projection: Y[n][0:3072] = x_row[n] @ [W_izr;W_it]^T + bc,  n = b*16+t
// Block tile 256(M) x 128(N); 8 waves as 4(M) x 2(N); wave tile 64x64.
// Row-contiguous Y => epilogue stores use one base + immediate offsets.
// ---------------------------------------------------------------------------
__global__ __launch_bounds__(256) void gemm_xproj(
    const bf16_t* __restrict__ Xbf,   // [65536][1024]
    const bf16_t* __restrict__ Wc,    // [3072][1024] rows: 0..2047=W_izr, 2048..=W_it
    const float* __restrict__ bc,     // [3072] concat(b_izr, b_it)
    bf16_t* __restrict__ Y)           // [65536][3072]
{
  __shared__ __align__(16) bf16_t Xs[256 * LDSP];
  __shared__ __align__(16) bf16_t Ws[128 * LDSP];
  const int tid = threadIdx.x;
  const int wave = tid >> 5, lane = tid & 31;
  const int hl = lane >> 4, l16 = lane & 15;
  const int wm = wave & 3, wn = wave >> 2;
  const int row0 = blockIdx.y * 256;
  const int col0 = blockIdx.x * 128;

  v8f zero = {0.f, 0.f, 0.f, 0.f, 0.f, 0.f, 0.f, 0.f};
  v8f acc[4][4];
#pragma unroll
  for (int mr = 0; mr < 4; ++mr)
#pragma unroll
    for (int nf = 0; nf < 4; ++nf) acc[mr][nf] = zero;

  for (int k0 = 0; k0 < 1024; k0 += 32) {
#pragma unroll
    for (int i = 0; i < 4; ++i) {
      int u = tid + 256 * i;
      int r = u >> 2;
      int c = (u & 3) << 3;
      *(uint4*)&Xs[r * LDSP + c] =
          *(const uint4*)&Xbf[(size_t)(row0 + r) * 1024 + (k0 + c)];
    }
#pragma unroll
    for (int i = 0; i < 2; ++i) {
      int u = tid + 256 * i;
      int r = u >> 2;
      int c = (u & 3) << 3;
      *(uint4*)&Ws[r * LDSP + c] =
          *(const uint4*)&Wc[(size_t)(col0 + r) * 1024 + (k0 + c)];
    }
    __syncthreads();

    FragU a[4], b[4];
#pragma unroll
    for (int mr = 0; mr < 4; ++mr) {
      const bf16_t* rp = &Xs[(wm * 64 + mr * 16 + l16) * LDSP];
      a[mr].u[0] = *(const uint4*)&rp[hl * 8];
      a[mr].u[1] = *(const uint4*)&rp[16 + hl * 8];
    }
#pragma unroll
    for (int nf = 0; nf < 4; ++nf) {
      const bf16_t* rp = &Ws[(wn * 64 + nf * 16 + l16) * LDSP];
      b[nf].u[0] = *(const uint4*)&rp[hl * 8];
      b[nf].u[1] = *(const uint4*)&rp[16 + hl * 8];
    }
#pragma unroll
    for (int mr = 0; mr < 4; ++mr)
#pragma unroll
      for (int nf = 0; nf < 4; ++nf)
        acc[mr][nf] = __builtin_amdgcn_wmma_f32_16x16x32_bf16(
            false, a[mr].v, false, b[nf].v, (short)0, acc[mr][nf], false, false);
    __syncthreads();
  }

  // epilogue: + bias, contiguous row stores (stride 3072 elems per v fits imm offset)
#pragma unroll
  for (int nf = 0; nf < 4; ++nf) {
    const int col = col0 + wn * 64 + nf * 16 + l16;
    const float bv = bc[col];
#pragma unroll
    for (int mr = 0; mr < 4; ++mr) {
      const int n0 = row0 + wm * 64 + mr * 16 + 8 * hl;   // rows n0..n0+7
      bf16_t* yp = Y + (size_t)n0 * 3072 + col;
#pragma unroll
      for (int v = 0; v < 8; ++v)
        yp[v * 3072] = (bf16_t)(acc[mr][nf][v] + bv);
    }
  }
}

// ---------------------------------------------------------------------------
// One GRU timestep, both directions (blockIdx.z). Block 128 rows x 128 cols;
// 8 waves as 2(M) x 4(J); wave tile 64x32 per gate = 24 WMMAs per K-chunk.
// ---------------------------------------------------------------------------
__global__ __launch_bounds__(256) void gru_step(
    const bf16_t* __restrict__ Whc,   // [3072][1024] rows: 0..1023 r, 1024..2047 z, 2048.. cand
    const float* __restrict__ b_hzr, const float* __restrict__ b_ht,
    const bf16_t* __restrict__ Y,     // [65536][3072], row n = b*16+t
    const float* __restrict__ hfin, const bf16_t* __restrict__ hfin_bf,
    float* __restrict__ hfout, bf16_t* __restrict__ hfout_bf,
    const float* __restrict__ hbin, const bf16_t* __restrict__ hbin_bf,
    float* __restrict__ hbout, bf16_t* __restrict__ hbout_bf,
    float* __restrict__ out, int t_f, int t_b, int tout_f, int tout_b)
{
  const float* hin; const bf16_t* hinbf;
  float* hout; bf16_t* houtbf; int t_in, tout, doff;
  if (blockIdx.z == 0) {
    hin = hfin; hinbf = hfin_bf; hout = hfout; houtbf = hfout_bf;
    t_in = t_f; tout = tout_f; doff = 0;
  } else {
    hin = hbin; hinbf = hbin_bf; hout = hbout; houtbf = hbout_bf;
    t_in = t_b; tout = tout_b; doff = 1024;
  }

  __shared__ __align__(16) bf16_t Hs[128 * LDSP];
  const int tid = threadIdx.x;
  const int wave = tid >> 5, lane = tid & 31;
  const int hl = lane >> 4, l16 = lane & 15;
  const int wm = wave & 1;       // 0..1 -> 64-row subtile
  const int wj = wave >> 1;      // 0..3 -> 32-col subtile
  const int row0 = blockIdx.y * 128;
  const int j0 = blockIdx.x * 128;

  v8f zero = {0.f, 0.f, 0.f, 0.f, 0.f, 0.f, 0.f, 0.f};
  v8f acc[3][4][2];
#pragma unroll
  for (int g = 0; g < 3; ++g)
#pragma unroll
    for (int mr = 0; mr < 4; ++mr)
#pragma unroll
      for (int jf = 0; jf < 2; ++jf) acc[g][mr][jf] = zero;

  for (int k0 = 0; k0 < 1024; k0 += 32) {
#pragma unroll
    for (int i = 0; i < 2; ++i) {
      int u = tid + 256 * i;
      int r = u >> 2;
      int c = (u & 3) << 3;
      *(uint4*)&Hs[r * LDSP + c] =
          *(const uint4*)&hinbf[(size_t)(row0 + r) * 1024 + (k0 + c)];
    }
    __syncthreads();

    FragU a[4];
#pragma unroll
    for (int mr = 0; mr < 4; ++mr) {
      const bf16_t* rp = &Hs[(wm * 64 + mr * 16 + l16) * LDSP];
      a[mr].u[0] = *(const uint4*)&rp[hl * 8];
      a[mr].u[1] = *(const uint4*)&rp[16 + hl * 8];
    }
    FragU b[3][2];
#pragma unroll
    for (int g = 0; g < 3; ++g)
#pragma unroll
      for (int jf = 0; jf < 2; ++jf) {
        const bf16_t* rp =
            &Whc[(size_t)(g * 1024 + j0 + wj * 32 + jf * 16 + l16) * 1024 + k0];
        b[g][jf].u[0] = *(const uint4*)&rp[hl * 8];
        b[g][jf].u[1] = *(const uint4*)&rp[16 + hl * 8];
      }
#pragma unroll
    for (int g = 0; g < 3; ++g)
#pragma unroll
      for (int mr = 0; mr < 4; ++mr)
#pragma unroll
        for (int jf = 0; jf < 2; ++jf)
          acc[g][mr][jf] = __builtin_amdgcn_wmma_f32_16x16x32_bf16(
              false, a[mr].v, false, b[g][jf].v, (short)0, acc[g][mr][jf],
              false, false);
    __syncthreads();
  }

  // gates + state update + output store (hoisted bases, immediate-offset walks)
#pragma unroll
  for (int jf = 0; jf < 2; ++jf) {
    const int jg = j0 + wj * 32 + jf * 16 + l16;
    const float brv = b_hzr[jg];
    const float bzv = b_hzr[1024 + jg];
    const float btv = b_ht[jg];
#pragma unroll
    for (int mr = 0; mr < 4; ++mr) {
      const int r0 = row0 + wm * 64 + mr * 16 + 8 * hl;   // batch rows r0..r0+7
      const bf16_t* yp  = Y + ((size_t)r0 * 16 + t_in) * 3072 + jg;   // stride 16*3072/v
      const float*  hip = hin  + (size_t)r0 * 1024 + jg;               // stride 1024/v
      float*        hop = hout + (size_t)r0 * 1024 + jg;
      bf16_t*       hbp = houtbf + (size_t)r0 * 1024 + jg;
      float*        op  = out + ((size_t)r0 * 16 + tout) * 2048 + doff + jg; // stride 16*2048/v
#pragma unroll
      for (int v = 0; v < 8; ++v) {
        const float rpre = acc[0][mr][jf][v] + brv + (float)yp[v * 49152];
        const float zpre = acc[1][mr][jf][v] + bzv + (float)yp[v * 49152 + 1024];
        const float r = sigmoidf_(rpre);
        const float z = sigmoidf_(zpre);
        const float cand =
            tanhf_((float)yp[v * 49152 + 2048] + r * (acc[2][mr][jf][v] + btv));
        const float hold = hip[v * 1024];
        const float hn = z * cand + (1.0f - z) * hold;
        hop[v * 1024] = hn;
        hbp[v * 1024] = (bf16_t)hn;
        op[v * 32768] = hn;
      }
    }
  }
}

// ---------------------------------------------------------------------------
extern "C" void kernel_launch(void* const* d_in, const int* in_sizes, int n_in,
                              void* d_out, int out_size, void* d_ws, size_t ws_size,
                              hipStream_t stream) {
  (void)in_sizes; (void)n_in; (void)out_size; (void)ws_size;
  const float* x     = (const float*)d_in[0];
  const float* h0    = (const float*)d_in[1];
  const float* bih0  = (const float*)d_in[2];
  const float* W_izr = (const float*)d_in[3];
  const float* b_izr = (const float*)d_in[4];
  const float* W_hzr = (const float*)d_in[5];
  const float* b_hzr = (const float*)d_in[6];
  const float* W_it  = (const float*)d_in[7];
  const float* b_it  = (const float*)d_in[8];
  const float* W_ht  = (const float*)d_in[9];
  const float* b_ht  = (const float*)d_in[10];
  float* out = (float*)d_out;

  char* p = (char*)d_ws;
  bf16_t* Xbf = (bf16_t*)p; p += (size_t)65536 * 1024 * 2;      // 128 MB
  bf16_t* Wc  = (bf16_t*)p; p += (size_t)3072 * 1024 * 2;       // 6 MB
  bf16_t* Whc = (bf16_t*)p; p += (size_t)3072 * 1024 * 2;       // 6 MB
  bf16_t* Y   = (bf16_t*)p; p += (size_t)65536 * 3072 * 2;      // 384 MB
  float*  bc  = (float*)p;  p += (size_t)3072 * 4;              // 12 KB
  float*  hf32[2]; float* hb32[2]; bf16_t* hfbf[2]; bf16_t* hbbf[2];
  for (int i = 0; i < 2; ++i) { hf32[i] = (float*)p; p += (size_t)4096 * 1024 * 4; }
  for (int i = 0; i < 2; ++i) { hb32[i] = (float*)p; p += (size_t)4096 * 1024 * 4; }
  for (int i = 0; i < 2; ++i) { hfbf[i] = (bf16_t*)p; p += (size_t)4096 * 1024 * 2; }
  for (int i = 0; i < 2; ++i) { hbbf[i] = (bf16_t*)p; p += (size_t)4096 * 1024 * 2; }

  auto cvt = [&](const float* src, bf16_t* dst, long long n) {
    long long n8 = n >> 3;
    cvt_f32_bf16<<<dim3((unsigned)((n8 + 255) / 256)), 256, 0, stream>>>(src, dst, n8);
  };
  cvt(x, Xbf, (long long)65536 * 1024);
  cvt(W_izr, Wc, (long long)2048 * 1024);
  cvt(W_it, Wc + (size_t)2048 * 1024, (long long)1024 * 1024);
  cvt(W_hzr, Whc, (long long)2048 * 1024);
  cvt(W_ht, Whc + (size_t)2048 * 1024, (long long)1024 * 1024);
  cvt(h0, hfbf[1], (long long)4096 * 1024);    // step 0 reads parity-1 buffer
  cvt(bih0, hbbf[1], (long long)4096 * 1024);
  copy_f32<<<dim3(8), 256, 0, stream>>>(b_izr, bc, 2048);
  copy_f32<<<dim3(4), 256, 0, stream>>>(b_it, bc + 2048, 1024);

  gemm_xproj<<<dim3(24, 256), 256, 0, stream>>>(Xbf, Wc, bc, Y);

  for (int s = 0; s < 16; ++s) {
    const float*  hf_in   = (s == 0) ? h0   : hf32[(s + 1) & 1];
    const float*  hb_in   = (s == 0) ? bih0 : hb32[(s + 1) & 1];
    const bf16_t* hf_inbf = hfbf[(s + 1) & 1];
    const bf16_t* hb_inbf = hbbf[(s + 1) & 1];
    gru_step<<<dim3(8, 32, 2), 256, 0, stream>>>(
        Whc, b_hzr, b_ht, Y,
        hf_in, hf_inbf, hf32[s & 1], hfbf[s & 1],
        hb_in, hb_inbf, hb32[s & 1], hbbf[s & 1],
        out,
        /*t_f=*/s, /*t_b=*/15 - s,           // time index consumed
        /*tout_f=*/15 - s, /*tout_b=*/s);    // output time slot
  }
}
